// Encoder_39204461478143
// MI455X (gfx1250) — matrix operand order repaired and checked by reading, not compiled, not run on hardware
//
#include <hip/hip_runtime.h>
#include <hip/hip_bf16.h>
#include <stdint.h>
#include <math.h>

// ---------------- problem dims ----------------
#define T_LEN 512
#define BATCH 64
#define EDIM  512
#define HDIM  512
#define GDIM  2048                 // 4*H
#define ROWS  (T_LEN * BATCH)      // 32768

// ---------------- WMMA types ----------------
typedef __attribute__((ext_vector_type(16))) __bf16 v16bf;
typedef __attribute__((ext_vector_type(8)))  float  v8f;

union FragU { uint4 q[2]; v16bf v; };

__device__ __forceinline__ unsigned short f2bf(float f) {
    union { float f; unsigned u; } v; v.f = f;
    unsigned u = v.u;
    unsigned r = u + 0x7FFFu + ((u >> 16) & 1u);   // round-to-nearest-even
    return (unsigned short)(r >> 16);
}

// A fragment (16xK tile of bf16, row-major source, ldk halves per row).
// ISA 7.12.2: lane L (L<16): M=L, K 0..7 in v0..3, K 16..23 in v4..7;
// lanes 16..31: M=L-16, K 8..15 / 24..31.
__device__ __forceinline__ v16bf load_a_frag(const unsigned short* base, int ldk,
                                             int mbase, int ktile, int lane) {
    const int row = mbase + (lane & 15);
    const int hi  = lane >> 4;
    const unsigned short* p = base + (size_t)row * ldk;
    FragU u;
    u.q[0] = *(const uint4*)(p + ktile + hi * 8);
    u.q[1] = *(const uint4*)(p + ktile + 16 + hi * 8);
    return u.v;
}

// B fragment (Kx16 tile) from an LDS-resident weight panel laid out
// [col][ldk] halves (K contiguous per output column).
// lanes 0..15 hold K 0..7 / 16..23 of col n=lane, lanes 16..31 hold K 8..15 / 24..31.
__device__ __forceinline__ v16bf lds_b_frag(const unsigned short* Wt, int ldk,
                                            int colbase, int ktile, int lane) {
    const int col = colbase + (lane & 15);
    const int hi  = lane >> 4;
    const unsigned short* p = Wt + (size_t)col * ldk + ktile + hi * 8;
    FragU u;
    u.q[0] = *(const uint4*)(p);
    u.q[1] = *(const uint4*)(p + 16);
    return u.v;
}

__device__ __forceinline__ float sigm(float x) { return 1.0f / (1.0f + __expf(-x)); }

// ---------------- kernel 0: zero state region ----------------
__global__ void zero_kernel(uint4* p, int n16) {
    int i = blockIdx.x * blockDim.x + threadIdx.x;
    if (i < n16) p[i] = (uint4){0u, 0u, 0u, 0u};
}

// ---------------- kernel 1: embedding gather -> bf16 ----------------
__global__ void embed_gather(const int* __restrict__ seq,
                             const float* __restrict__ emb,
                             unsigned short* __restrict__ xbf) {
    int gid = blockIdx.x * blockDim.x + threadIdx.x;    // ROWS*EDIM/4 threads
    int row = gid >> 7;                                  // 128 quads per row
    int e4  = (gid & 127) << 2;
    int tok = seq[row];
    float4 v = *(const float4*)(emb + (size_t)tok * EDIM + e4);
    uint2 pk;
    pk.x = (unsigned)f2bf(v.x) | ((unsigned)f2bf(v.y) << 16);
    pk.y = (unsigned)f2bf(v.z) | ((unsigned)f2bf(v.w) << 16);
    *(uint2*)(xbf + (size_t)row * EDIM + e4) = pk;
}

// ---------------- kernel 2: f32 -> bf16 weight convert ----------------
__global__ void cvt_bf16(const float* __restrict__ src,
                         unsigned short* __restrict__ dst, int n4) {
    int gid = blockIdx.x * blockDim.x + threadIdx.x;
    if (gid >= n4) return;
    float4 v = ((const float4*)src)[gid];
    uint2 pk;
    pk.x = (unsigned)f2bf(v.x) | ((unsigned)f2bf(v.y) << 16);
    pk.y = (unsigned)f2bf(v.z) | ((unsigned)f2bf(v.w) << 16);
    ((uint2*)dst)[gid] = pk;
}

// ---------------- kernel 3: input-projection GEMM (both directions) ------
// xproj[dir][row, g] = sum_e x[row,e] * Wih[dir][g,e] + b_ih[g] + b_hh[g]
// Block: 8 waves x (16 rows each) x 64 cols. Weight panel staged in LDS in
// two 32KB K-chunks (K=256 halves) shared by all 8 waves.
__global__ void __launch_bounds__(256) xproj_gemm(
        const unsigned short* __restrict__ xbf,
        const unsigned short* __restrict__ WihL,
        const unsigned short* __restrict__ WihR,
        const float* __restrict__ bihL, const float* __restrict__ bhhL,
        const float* __restrict__ bihR, const float* __restrict__ bhhR,
        float* __restrict__ xpL, float* __restrict__ xpR) {
    __shared__ unsigned short Wtile[64 * 256];   // 32 KB panel: [col][256]

    const int lane = threadIdx.x & 31;
    const int wid  = threadIdx.x >> 5;

    // blocks: 2 dirs x 256 row-groups x 32 col-groups = 16384
    const int per_dir = (ROWS / 128) * (GDIM / 64);   // 256*32 = 8192
    const int dir = blockIdx.x / per_dir;
    const int rem = blockIdx.x % per_dir;
    const int mg  = rem % (ROWS / 128);               // row group (128 rows)
    const int n64 = rem / (ROWS / 128);               // col group (64 cols)

    const unsigned short* W = dir ? WihR : WihL;
    const float* bih = dir ? bihR : bihL;
    const float* bhh = dir ? bhhR : bhhL;
    float* xp        = dir ? xpR  : xpL;

    const int mbase = (mg * 8 + wid) * 16;            // this wave's 16 rows
    const int nb0   = n64 * 64;
    const int ncol  = lane & 15;
    const int hi    = lane >> 4;

    v8f acc[4];
#pragma unroll
    for (int j = 0; j < 4; ++j) {
        float b = bih[nb0 + j * 16 + ncol] + bhh[nb0 + j * 16 + ncol];
#pragma unroll
        for (int r = 0; r < 8; ++r) acc[j][r] = b;
    }

    for (int c = 0; c < 2; ++c) {                     // two K chunks of 256
        const int k0 = c * 256;
        // cooperative copy: 64 cols x 256 halves = 2048 uint4, 8 per thread
        for (int i = threadIdx.x; i < 2048; i += 256) {
            const int col = i >> 5;                   // 32 uint4 per col
            const int q   = i & 31;
            ((uint4*)Wtile)[i] =
                *(const uint4*)(W + (size_t)(nb0 + col) * EDIM + k0 + q * 8);
        }
        __syncthreads();

        for (int kl = 0; kl < 256; kl += 32) {
            if (c == 0)   // speculative prefetch of this row's next-chunk A data
                __builtin_prefetch((const void*)(xbf + (size_t)(mbase + ncol) * EDIM + 256 + kl), 0, 1);
            v16bf a = load_a_frag(xbf, EDIM, mbase, k0 + kl, lane);
#pragma unroll
            for (int j = 0; j < 4; ++j) {
                v16bf bm = lds_b_frag(Wtile, 256, j * 16, kl, lane);
                acc[j] = __builtin_amdgcn_wmma_f32_16x16x32_bf16(
                    false, a, false, bm, (short)0, acc[j], false, false);
            }
        }
        __syncthreads();
    }

#pragma unroll
    for (int j = 0; j < 4; ++j)
#pragma unroll
        for (int r = 0; r < 8; ++r) {
            int row = mbase + r + 8 * hi;
            xp[(size_t)row * GDIM + nb0 + j * 16 + ncol] = acc[j][r];
        }
}

// ---------------- persistent bidirectional LSTM scan ----------------
__device__ __forceinline__ void grid_barrier(unsigned* ctr, unsigned phase) {
    __threadfence();
    __syncthreads();
    if (threadIdx.x == 0) {
        __hip_atomic_fetch_add(ctr, 1u, __ATOMIC_ACQ_REL, __HIP_MEMORY_SCOPE_AGENT);
        const unsigned target = phase * 64u;
        while (__hip_atomic_load(ctr, __ATOMIC_ACQUIRE, __HIP_MEMORY_SCOPE_AGENT) < target)
            __builtin_amdgcn_s_sleep(2);
    }
    __syncthreads();
}

__global__ void __launch_bounds__(256) bilstm_scan(
        const float* __restrict__ xpL, const float* __restrict__ xpR,
        const unsigned short* __restrict__ WhhL,
        const unsigned short* __restrict__ WhhR,
        float* gates,          // [2][64][2048] scratch
        unsigned short* hb,    // [2][64][512]  bf16 hidden state
        float* cst,            // [2][64][512]  f32 cell state
        float* out,            // [512][64][1024]
        unsigned* ctr) {       // [2] barrier counters (zeroed before launch)
    // This block's 32 W_hh columns, resident in LDS for ALL 512 timesteps.
    __shared__ unsigned short Wtile[32 * 512];   // 32 KB: [col][512]

    const int dir  = blockIdx.x >> 6;       // 0: forward, 1: reverse
    const int lb   = blockIdx.x & 63;       // block within direction
    const int lane = threadIdx.x & 31;
    const int wid  = threadIdx.x >> 5;

    const float* xp          = dir ? xpR  : xpL;
    const unsigned short* W  = dir ? WhhR : WhhL;
    float* g         = gates + (size_t)dir * BATCH * GDIM;
    unsigned short* h = hb   + (size_t)dir * BATCH * HDIM;
    float* c         = cst   + (size_t)dir * BATCH * HDIM;
    unsigned* cc     = ctr + dir;

    const int wgl   = lb * 8 + wid;   // 0..511 tiles: 4 mtiles x 128 ntiles
    const int mbase = (wgl & 3) * 16; // batch tile
    const int nbase = (wgl >> 2) * 16;
    const int lcolb = ((wgl >> 2) & 1) * 16;   // column base within LDS panel
    const int ncol  = lane & 15;
    const int hi    = lane >> 4;

    // one-time W_hh panel load: cols [32*lb, 32*lb+32), 2048 uint4
    for (int i = threadIdx.x; i < 2048; i += 256) {
        const int col = i >> 6;                    // 64 uint4 per col of 512
        const int q   = i & 63;
        ((uint4*)Wtile)[i] =
            *(const uint4*)(W + (size_t)(lb * 32 + col) * HDIM + q * 8);
    }
    __syncthreads();

    unsigned phase = 0;

    for (int s = 0; s < T_LEN; ++s) {
        const int t = dir ? (T_LEN - 1 - s) : s;

        // ---- phase A: gates = xproj[t] + h_prev @ Whh^T (WMMA) ----
        const float* xpt = xp + (size_t)t * BATCH * GDIM;
        v8f acc;
#pragma unroll
        for (int r = 0; r < 8; ++r)
            acc[r] = xpt[(size_t)(mbase + r + 8 * hi) * GDIM + nbase + ncol];

        for (int k = 0; k < HDIM; k += 32) {
            v16bf a  = load_a_frag(h, HDIM, mbase, k, lane);
            v16bf bm = lds_b_frag(Wtile, 512, lcolb, k, lane);
            acc = __builtin_amdgcn_wmma_f32_16x16x32_bf16(
                false, a, false, bm, (short)0, acc, false, false);
        }
#pragma unroll
        for (int r = 0; r < 8; ++r)
            g[(size_t)(mbase + r + 8 * hi) * GDIM + nbase + ncol] = acc[r];

        grid_barrier(cc, ++phase);

        // ---- phase B: fused LSTM cell (i,f,g,o in PyTorch order) ----
        const int base = (lb * 256 + threadIdx.x) * 2;   // 32768 elems / 16384 threads
#pragma unroll
        for (int e = 0; e < 2; ++e) {
            const int idx = base + e;
            const int b   = idx >> 9;      // / 512
            const int hu  = idx & 511;
            const float gi = g[(size_t)b * GDIM + hu];
            const float gf = g[(size_t)b * GDIM + 512 + hu];
            const float gc = g[(size_t)b * GDIM + 1024 + hu];
            const float go = g[(size_t)b * GDIM + 1536 + hu];
            const float co = c[(size_t)b * HDIM + hu];
            const float cn = sigm(gf) * co + sigm(gi) * tanhf(gc);
            const float hn = sigm(go) * tanhf(cn);
            c[(size_t)b * HDIM + hu] = cn;
            h[(size_t)b * HDIM + hu] = f2bf(hn);
            out[((size_t)t * BATCH + b) * (2 * HDIM) + dir * HDIM + hu] = hn;
        }
        grid_barrier(cc, ++phase);
    }
}

// ---------------- host side ----------------
extern "C" void kernel_launch(void* const* d_in, const int* in_sizes, int n_in,
                              void* d_out, int out_size, void* d_ws, size_t ws_size,
                              hipStream_t stream) {
    (void)in_sizes; (void)n_in; (void)out_size; (void)ws_size;

    const int*   seq  = (const int*)d_in[0];
    const float* emb  = (const float*)d_in[1];
    const float* WihL = (const float*)d_in[2];
    const float* WhhL = (const float*)d_in[3];
    const float* bihL = (const float*)d_in[4];
    const float* bhhL = (const float*)d_in[5];
    const float* WihR = (const float*)d_in[6];
    const float* WhhR = (const float*)d_in[7];
    const float* bihR = (const float*)d_in[8];
    const float* bhhR = (const float*)d_in[9];
    float* out = (float*)d_out;

    char* ws = (char*)d_ws;
    size_t off = 0;
    auto alloc = [&](size_t bytes) -> char* {
        char* p = ws + off;
        off += (bytes + 255) & ~(size_t)255;
        return p;
    };

    unsigned short* xbf   = (unsigned short*)alloc((size_t)ROWS * EDIM * 2);
    unsigned short* wihLb = (unsigned short*)alloc((size_t)GDIM * EDIM * 2);
    unsigned short* wihRb = (unsigned short*)alloc((size_t)GDIM * EDIM * 2);
    unsigned short* whhLb = (unsigned short*)alloc((size_t)GDIM * HDIM * 2);
    unsigned short* whhRb = (unsigned short*)alloc((size_t)GDIM * HDIM * 2);
    float* xpL = (float*)alloc((size_t)ROWS * GDIM * 4);
    float* xpR = (float*)alloc((size_t)ROWS * GDIM * 4);
    // contiguous state region (zeroed every call for determinism)
    char* state = ws + off;
    float*          gates = (float*)alloc((size_t)2 * BATCH * GDIM * 4);
    unsigned short* hb    = (unsigned short*)alloc((size_t)2 * BATCH * HDIM * 2);
    float*          cst   = (float*)alloc((size_t)2 * BATCH * HDIM * 4);
    unsigned*       ctr   = (unsigned*)alloc(256);
    const size_t stateBytes = (size_t)((ws + off) - state);

    // 0) zero barrier counters + h/c state (+ gates, harmless)
    {
        int n16 = (int)(stateBytes / 16);
        zero_kernel<<<(n16 + 255) / 256, 256, 0, stream>>>((uint4*)state, n16);
    }
    // 1) embedding gather -> bf16
    embed_gather<<<(ROWS * EDIM / 4) / 256, 256, 0, stream>>>(seq, emb, xbf);
    // 2) weights -> bf16
    {
        const int n4 = GDIM * EDIM / 4;
        const int nb = (n4 + 255) / 256;
        cvt_bf16<<<nb, 256, 0, stream>>>(WihL, wihLb, n4);
        cvt_bf16<<<nb, 256, 0, stream>>>(WihR, wihRb, n4);
        cvt_bf16<<<nb, 256, 0, stream>>>(WhhL, whhLb, n4);
        cvt_bf16<<<nb, 256, 0, stream>>>(WhhR, whhRb, n4);
    }
    // 3) input projection GEMM, both directions (16384 blocks x 8 waves)
    xproj_gemm<<<16384, 256, 0, stream>>>(xbf, wihLb, wihRb,
                                          bihL, bhhL, bihR, bhhR, xpL, xpR);
    // 4) persistent bidirectional LSTM scan (64 WGs per direction)
    bilstm_scan<<<128, 256, 0, stream>>>(xpL, xpR, whhLb, whhRb,
                                         gates, hb, cst, out, ctr);
}
